// SelfAttentionHead_38766374813843
// MI455X (gfx1250) — matrix operand order repaired and checked by reading, not compiled, not run on hardware
//
#include <hip/hip_runtime.h>
#include <hip/hip_bf16.h>

typedef __attribute__((ext_vector_type(16))) __bf16       v16bf;
typedef __attribute__((ext_vector_type(8)))  float        v8f;
typedef __attribute__((ext_vector_type(8)))  unsigned int v8u;

#define HEAD_DIM 64
#define EMBED    1024
#define TSEQ     4096
#define NBATCH   4
#define NROWS    (NBATCH * TSEQ)   // 16384

// f32 -> bf16 (single): let the compiler pick the native cvt for gfx1250
__device__ __forceinline__ unsigned short bf16s(float f) {
    return __builtin_bit_cast(unsigned short, (__bf16)f);
}

// pack two f32 into one dword of 2x bf16 (lo | hi<<16)
__device__ __forceinline__ unsigned int pkbf(float lo, float hi) {
#if __has_builtin(__builtin_amdgcn_cvt_pk_bf16_f32)
    auto r = __builtin_amdgcn_cvt_pk_bf16_f32(lo, hi);   // v_cvt_pk_bf16_f32
    return __builtin_bit_cast(unsigned int, r);
#else
    return (unsigned int)bf16s(lo) | ((unsigned int)bf16s(hi) << 16);
#endif
}
__device__ __forceinline__ v16bf as_bf(v8u u) {
    return __builtin_bit_cast(v16bf, u);
}

// ---------------------------------------------------------------------------
// Projection: [16384,1024] x [1024,64] -> Q,K (bf16 row-major [row][64]) and
// V transposed (bf16 Vt[b][64][4096]).  One wave = one 16x16 output tile.
// grid: 3072 blocks = 1024 rowtiles x 3 mats (block-uniform mat -> scalar
// branches for the store path); 4 waves per block = 4 column tiles.
// ---------------------------------------------------------------------------
__global__ void __launch_bounds__(128)
proj_kernel(const float* __restrict__ x,
            const float* __restrict__ Wq,
            const float* __restrict__ Wk,
            const float* __restrict__ Wv,
            unsigned short* __restrict__ Qbf,
            unsigned short* __restrict__ Kbf,
            unsigned short* __restrict__ Vt)
{
    const int lane = threadIdx.x & 31;
    const int wave = threadIdx.x >> 5;
    const int mat  = blockIdx.x % 3;     // uniform per block: 0=Q 1=K 2=V
    const int mt   = blockIdx.x / 3;     // row tile (of 1024)
    const int nt   = wave;               // column tile (of 4)
    const int m0   = mt * 16;
    const int n0   = nt * 16;
    const int h    = lane >> 4;
    const int lm   = lane & 15;

    const float* W = (mat == 0) ? Wq : (mat == 1) ? Wk : Wv;

    v8f acc = {};
    const float* xr = x + (size_t)(m0 + lm) * EMBED;

    for (int kc = 0; kc < EMBED; kc += 32) {
        // A fragment: x rows m0..m0+15, K = kc..kc+31 (contiguous k order: k = kc + 16*h + s)
        v8u au;
#pragma unroll
        for (int j = 0; j < 8; ++j) {
            int d = kc + 16 * h + 2 * j;
            au[j] = pkbf(xr[d], xr[d + 1]);
        }
        // B fragment: W[kc..kc+31][n0..n0+15]; lane halves split K 0..15 / 16..31
        v8u bu;
#pragma unroll
        for (int j = 0; j < 8; ++j) {
            int k = kc + 16 * h + 2 * j;
            bu[j] = pkbf(W[(size_t)k * HEAD_DIM + n0 + lm],
                         W[(size_t)(k + 1) * HEAD_DIM + n0 + lm]);
        }
        acc = __builtin_amdgcn_wmma_f32_16x16x32_bf16(
                  false, as_bf(au), false, as_bf(bu), (short)0, acc, false, false);
    }

    // C/D layout: element j -> row = j + 8*h, col = n0 + lm
    if (mat != 2) {
        unsigned short* dst = (mat == 0) ? Qbf : Kbf;
#pragma unroll
        for (int j = 0; j < 8; ++j) {
            int row = m0 + j + 8 * h;
            dst[(size_t)row * HEAD_DIM + n0 + lm] = bf16s(acc[j]);
        }
    } else {
#pragma unroll
        for (int j = 0; j < 8; ++j) {
            int row = m0 + j + 8 * h;
            int bb  = row >> 12;          // / 4096
            int t   = row & (TSEQ - 1);
            Vt[((size_t)(bb * HEAD_DIM + n0 + lm)) * TSEQ + t] = bf16s(acc[j]);
        }
    }
}

// ---------------------------------------------------------------------------
// Flash attention: one wave owns 16 query rows; 32 keys per step.
// S via 4 WMMAs (two 16x16 subtiles, K=64 over head dim), online softmax with
// shfl_xor row reductions, P restaged through per-wave LDS (bf16) into an
// A-fragment, P*V via 4 WMMAs (N=64).  Next K/V tile prefetched each step.
// ---------------------------------------------------------------------------
__global__ void __launch_bounds__(128)
attn_kernel(const unsigned short* __restrict__ Qbf,
            const unsigned short* __restrict__ Kbf,
            const unsigned short* __restrict__ Vt,
            float* __restrict__ out)
{
    __shared__ unsigned int PlU[4][256];   // per-wave 16x32 bf16 P tile (1KB each)
    const int lane = threadIdx.x & 31;
    const int wave = threadIdx.x >> 5;
    const int tile = blockIdx.x * (blockDim.x >> 5) + wave;   // 0..1023
    const int b    = tile >> 8;          // / 256
    const int mt   = tile & 255;
    const int m0   = mt * 16;
    const int gq   = b * TSEQ + m0;      // global query-row base
    const int h    = lane >> 4;
    const int lm   = lane & 15;

    unsigned short* P = (unsigned short*)&PlU[wave][0];

    // Q A-fragments for d = 0..31 and 32..63
    v8u qa0, qa1;
    const unsigned short* qr = Qbf + (size_t)(gq + lm) * HEAD_DIM;
#pragma unroll
    for (int j = 0; j < 8; ++j) {
        int d = 16 * h + 2 * j;
        qa0[j] = *(const unsigned int*)(qr + d);
        qa1[j] = *(const unsigned int*)(qr + d + 32);
    }

    float m_st[8], l_st[8];
    v8f o0 = {}, o1 = {}, o2 = {}, o3 = {};
#pragma unroll
    for (int j = 0; j < 8; ++j) { m_st[j] = -1e30f; l_st[j] = 0.0f; }

    const float scale = 0.125f;          // 1/sqrt(64)
    const int n_end = m0 + 16;           // causal: keys < n_end

    for (int kb = 0; kb < n_end; kb += 32) {
        // ---- prefetch next K/V tile (lane-spread; emits global_prefetch) ----
        if (kb + 32 < n_end) {
            __builtin_prefetch(Kbf + (size_t)(b * TSEQ + kb + 32 + lane) * HEAD_DIM, 0, 3);
            __builtin_prefetch(Vt + ((size_t)(b * HEAD_DIM + lane)) * TSEQ + kb + 32, 0, 3);
            __builtin_prefetch(Vt + ((size_t)(b * HEAD_DIM + 32 + lane)) * TSEQ + kb + 32, 0, 3);
        }

        // ---- S = Q K^T : two 16x16 subtiles, K-dim 64 in two WMMA steps ----
        v8f s0 = {}, s1 = {};
        const unsigned short* kr = Kbf + (size_t)(b * TSEQ + kb + lm) * HEAD_DIM;
#pragma unroll
        for (int half = 0; half < 2; ++half) {
            v8u kf0, kf1;
#pragma unroll
            for (int j = 0; j < 8; ++j) {
                int d = half * 32 + 16 * h + 2 * j;
                kf0[j] = *(const unsigned int*)(kr + d);
                kf1[j] = *(const unsigned int*)(kr + 16 * HEAD_DIM + d);
            }
            v16bf qa = as_bf(half ? qa1 : qa0);
            s0 = __builtin_amdgcn_wmma_f32_16x16x32_bf16(
                     false, qa, false, as_bf(kf0), (short)0, s0, false, false);
            s1 = __builtin_amdgcn_wmma_f32_16x16x32_bf16(
                     false, qa, false, as_bf(kf1), (short)0, s1, false, false);
        }

        // ---- online softmax (rows live in 16-lane halves) ----
#pragma unroll
        for (int j = 0; j < 8; ++j) {
            int row = m0 + j + 8 * h;
            int k0  = kb + lm;
            int k1  = kb + 16 + lm;
            float a0 = (k0 <= row) ? s0[j] * scale : -1e30f;
            float a1 = (k1 <= row) ? s1[j] * scale : -1e30f;
            float rm = fmaxf(a0, a1);
#pragma unroll
            for (int msk = 1; msk < 16; msk <<= 1)
                rm = fmaxf(rm, __shfl_xor(rm, msk, 32));
            float mn = fmaxf(m_st[j], rm);
            float al = __expf(m_st[j] - mn);
            float p0 = __expf(a0 - mn);
            float p1 = __expf(a1 - mn);
            float rs = p0 + p1;
#pragma unroll
            for (int msk = 1; msk < 16; msk <<= 1)
                rs += __shfl_xor(rs, msk, 32);
            l_st[j] = l_st[j] * al + rs;
            m_st[j] = mn;
            o0[j] *= al; o1[j] *= al; o2[j] *= al; o3[j] *= al;
            int prow = j + 8 * h;
            P[prow * 32 + lm]      = bf16s(p0);
            P[prow * 32 + 16 + lm] = bf16s(p1);
        }

        // ---- P A-fragment from LDS (compiler inserts s_wait_dscnt) ----
        v8u pa;
#pragma unroll
        for (int j = 0; j < 8; ++j) {
            int k = 16 * h + 2 * j;
            pa[j] = *(const unsigned int*)(P + lm * 32 + k);
        }
        v16bf Pa = as_bf(pa);

        // ---- O += P * V  (4 N-tiles of 16; V transposed -> packed pairs) ----
#pragma unroll
        for (int nt = 0; nt < 4; ++nt) {
            v8u vb;
            const unsigned short* vr =
                Vt + ((size_t)(b * HEAD_DIM + nt * 16 + lm)) * TSEQ + kb + 16 * h;
#pragma unroll
            for (int j = 0; j < 8; ++j)
                vb[j] = *(const unsigned int*)(vr + 2 * j);
            v8f& o = (nt == 0) ? o0 : (nt == 1) ? o1 : (nt == 2) ? o2 : o3;
            o = __builtin_amdgcn_wmma_f32_16x16x32_bf16(
                    false, Pa, false, as_bf(vb), (short)0, o, false, false);
        }
    }

    // ---- finalize: out[row][d] = O / l ----
#pragma unroll
    for (int j = 0; j < 8; ++j) {
        float inv = 1.0f / l_st[j];
        float* orow = out + (size_t)(gq + j + 8 * h) * HEAD_DIM + lm;
        orow[0]  = o0[j] * inv;
        orow[16] = o1[j] * inv;
        orow[32] = o2[j] * inv;
        orow[48] = o3[j] * inv;
    }
}

// ---------------------------------------------------------------------------
extern "C" void kernel_launch(void* const* d_in, const int* in_sizes, int n_in,
                              void* d_out, int out_size, void* d_ws, size_t ws_size,
                              hipStream_t stream)
{
    const float* x  = (const float*)d_in[0];
    const float* Wq = (const float*)d_in[1];
    const float* Wk = (const float*)d_in[2];
    const float* Wv = (const float*)d_in[3];

    unsigned short* Qbf = (unsigned short*)d_ws;                       // 2 MB
    unsigned short* Kbf = Qbf + (size_t)NROWS * HEAD_DIM;              // 2 MB
    unsigned short* Vt  = Kbf + (size_t)NROWS * HEAD_DIM;              // 2 MB
    (void)ws_size; (void)in_sizes; (void)n_in; (void)out_size;

    // 1024 rowtiles x 3 mats -> 3072 blocks; 4 waves (coltiles) per block
    proj_kernel<<<3072, 128, 0, stream>>>(x, Wq, Wk, Wv, Qbf, Kbf, Vt);
    // 1024 query tiles, 4 waves per block -> 256 blocks
    attn_kernel<<<256, 128, 0, stream>>>(Qbf, Kbf, Vt, (float*)d_out);
}
